// HybridGNN_58196806861281
// MI455X (gfx1250) — matrix operand order repaired and checked by reading, not compiled, not run on hardware
//
#include <hip/hip_runtime.h>

#define B_     512
#define T_     2
#define D_     100
#define OUTD_  200
#define TOTALP_ 1086
#define NCOLS_ 112      // padded output-column space (7 tiles of 16)

typedef _Float16 v16h __attribute__((ext_vector_type(16)));
typedef _Float16 v8h  __attribute__((ext_vector_type(8)));
typedef float    v8f  __attribute__((ext_vector_type(8)));
typedef float    v4f  __attribute__((ext_vector_type(4)));

// Global-address-space views (force global_load/global_store, not flat).
typedef __attribute__((address_space(1))) float        gf1;
typedef __attribute__((address_space(1))) const float  gcf1;
typedef __attribute__((address_space(1))) const v4f    gcv4f1;

__device__ __forceinline__ v4f gload4(const float* p) {
  return *(const gcv4f1*)(unsigned long long)p;
}
__device__ __forceinline__ float gload1(const float* p) {
  return *(const gcf1*)(unsigned long long)p;
}
__device__ __forceinline__ void gstore1(float* p, float v) {
  *(gf1*)(unsigned long long)p = v;
}
// Clamped+masked float4 load: segment start s is a multiple of 4; a float4 is
// either fully inside [0,100) or fully invalid (-> zero), branchlessly.
__device__ __forceinline__ v4f gload4m(const float* p, int s) {
  int so = (s < D_) ? s : 0;
  v4f v = gload4(p + so);
  float msk = (s < D_) ? 1.f : 0.f;
  return v * msk;
}

// ---------------------------------------------------------------------------
// Row source descriptor (dense rows of 100 f32, or gather through index array)
// Parent row i: b = i/P, q = i%P; child j uses nb[b*bstride + toff + q*r + j].
// ---------------------------------------------------------------------------
struct GSrc {
  const float* dense;
  const float* table;
  const int*   nb;
  long long rowstride, coloff;
  int bstride, toff, P, r;
};

__device__ __forceinline__ v16h pack16(v4f f0, v4f f1, v4f f2, v4f f3) {
  v16h a;
  #pragma unroll
  for (int e = 0; e < 4; ++e) {
    a[e]      = (_Float16)f0[e];
    a[4 + e]  = (_Float16)f1[e];
    a[8 + e]  = (_Float16)f2[e];
    a[12 + e] = (_Float16)f3[e];
  }
  return a;
}

// ---------------------------------------------------------------------------
// Fused (gather + fanout-mean + GEMM + relu/residual) WMMA kernel.
//   out[n, N0+N1] = act( [X | mean_RY(Y)] @ blockdiag(W0[100xN0], W1[100xN1]) )
// RY = 0: plain GEMM (K=100).  RY in {3,5,7,9}: X at K=[0,100), mean(Y) at
// LDS rows [128,228) so contiguous A segments never straddle the two sources.
// 128 threads = 4 waves; each wave computes 16 rows x 100 cols (7 WMMA tiles).
// A fragment per 32-K step = two contiguous 8-float runs -> 4x b128 loads.
// B stored transposed in LDS ([112][KP+8] f16) -> 2x ds b128 per fragment.
// ---------------------------------------------------------------------------
template<int RY>
__global__ void wmma_gemm_kernel(GSrc X, GSrc Y,
                                 const float* __restrict__ W0, int N0,
                                 const float* __restrict__ W1, int N1,
                                 float* __restrict__ out, const float* __restrict__ resid,
                                 long long n, int doRelu)
{
  constexpr int KP    = (RY > 0) ? 256 : 128;
  constexpr int PITCH = KP + 8;              // f16 pitch: kills LDS bank conflicts
  extern __shared__ _Float16 BlsT[];         // [NCOLS_][PITCH]
  const int Ntot = N0 + N1;

  // Stage block-diagonal weights transposed into LDS (zero padded).
  for (int idx = threadIdx.x; idx < NCOLS_ * KP; idx += blockDim.x) {
    int nn = idx / KP;
    int k  = idx - nn * KP;
    float w = 0.f;
    if (k < D_) {
      if (nn < N0) w = W0[k * N0 + nn];
    } else if (RY > 0 && k >= 128 && k < 128 + D_) {
      if (nn >= N0 && nn < Ntot) w = W1[(k - 128) * N1 + (nn - N0)];
    }
    BlsT[nn * PITCH + k] = (_Float16)w;
  }
  __syncthreads();

  const int lane = threadIdx.x & 31;
  const int wave = threadIdx.x >> 5;
  const long long R = ((long long)blockIdx.x * (blockDim.x >> 5) + wave) * 16;
  if (R >= n) return;                        // wave-uniform exit

  const int m    = lane & 15;
  const int half = lane >> 4;
  long long row = R + m;
  if (row >= n) row = n - 1;                 // clamp; stores are guarded later

  // Resolve row pointers once (branch on .dense is wave-uniform).
  const float* xp;
  if (X.dense) {
    xp = X.dense + row * D_;
  } else {
    long long b = row / X.P;
    int q = (int)(row - b * X.P);
    int id = X.nb[b * (long long)X.bstride + X.toff + q];
    xp = X.table + (long long)id * X.rowstride + X.coloff;
  }
  const float* yp[RY > 0 ? RY : 1];
  if (RY > 0) {
    if (Y.dense) {
      #pragma unroll
      for (int j = 0; j < RY; ++j) yp[j] = Y.dense + (row * RY + j) * D_;
    } else {
      long long b = row / Y.P;
      int q = (int)(row - b * Y.P);
      const int* ip = Y.nb + b * (long long)Y.bstride + Y.toff + q * RY;
      #pragma unroll
      for (int j = 0; j < RY; ++j)
        yp[j] = Y.table + (long long)ip[j] * Y.rowstride + Y.coloff;
    }
  }
  const float invr = 1.f / (float)(RY > 0 ? RY : 1);

  v8f acc[7];
  #pragma unroll
  for (int c = 0; c < 7; ++c) {
    #pragma unroll
    for (int e = 0; e < 8; ++e) acc[c][e] = 0.f;
  }

  const _Float16* bbase = BlsT + half * 8 + m * PITCH;

  // ---- X phase: K = [0,128), A elements K>=100 are zero-masked ----
  #pragma unroll
  for (int k0 = 0; k0 < 128; k0 += 32) {
    const int s0 = k0 + half * 8;
    v4f f0 = gload4m(xp, s0);
    v4f f1 = gload4m(xp, s0 + 4);
    v4f f2 = gload4m(xp, s0 + 16);
    v4f f3 = gload4m(xp, s0 + 20);
    v16h a = pack16(f0, f1, f2, f3);
    #pragma unroll
    for (int c = 0; c < 7; ++c) {
      const _Float16* bp = bbase + c * (16 * PITCH) + k0;
      v8h blo = *(const v8h*)bp;
      v8h bhi = *(const v8h*)(bp + 16);
      v16h bf;
      #pragma unroll
      for (int e = 0; e < 8; ++e) { bf[e] = blo[e]; bf[8 + e] = bhi[e]; }
      acc[c] = __builtin_amdgcn_wmma_f32_16x16x32_f16(
          false, a, false, bf, (short)0, acc[c], false, false);
    }
  }

  // ---- Y phase: K = [128,256) maps to mean over RY children ----
  if (RY > 0) {
    #pragma unroll
    for (int k0 = 128; k0 < 256; k0 += 32) {
      const int s0 = (k0 - 128) + half * 8;
      v4f f0 = {0.f, 0.f, 0.f, 0.f};
      v4f f1 = f0, f2 = f0, f3 = f0;
      #pragma unroll
      for (int j = 0; j < RY; ++j) {
        f0 += gload4m(yp[j], s0);
        f1 += gload4m(yp[j], s0 + 4);
        f2 += gload4m(yp[j], s0 + 16);
        f3 += gload4m(yp[j], s0 + 20);
      }
      f0 *= invr; f1 *= invr; f2 *= invr; f3 *= invr;
      v16h a = pack16(f0, f1, f2, f3);
      #pragma unroll
      for (int c = 0; c < 7; ++c) {
        const _Float16* bp = bbase + c * (16 * PITCH) + k0;
        v8h blo = *(const v8h*)bp;
        v8h bhi = *(const v8h*)(bp + 16);
        v16h bf;
        #pragma unroll
        for (int e = 0; e < 8; ++e) { bf[e] = blo[e]; bf[8 + e] = bhi[e]; }
        acc[c] = __builtin_amdgcn_wmma_f32_16x16x32_f16(
            false, a, false, bf, (short)0, acc[c], false, false);
      }
    }
  }

  // C/D layout: M = v + 8*half, N = lane%16 + 16c
  #pragma unroll
  for (int c = 0; c < 7; ++c) {
    int ncol = c * 16 + m;
    if (ncol >= Ntot) continue;
    #pragma unroll
    for (int v = 0; v < 8; ++v) {
      long long rr = R + v + 8 * half;
      if (rr >= n) continue;
      float val = acc[c][v];
      if (doRelu) val = fmaxf(val, 0.f);
      long long o = rr * (long long)Ntot + ncol;
      if (resid) val += gload1(resid + o);
      gstore1(out + o, val);
    }
  }
}

// ---------------------------------------------------------------------------
// Row-wise LayerNorm, one wave32 per row of 100.
// ---------------------------------------------------------------------------
__global__ void ln_kernel(const float* __restrict__ x, float* __restrict__ y,
                          const float* __restrict__ g, const float* __restrict__ bb,
                          long long nrows)
{
  long long wid = (long long)blockIdx.x * (blockDim.x >> 5) + (threadIdx.x >> 5);
  int lane = threadIdx.x & 31;
  if (wid >= nrows) return;
  const float* xr = x + wid * D_;
  float vals[4]; float s = 0.f;
  #pragma unroll
  for (int j = 0; j < 4; ++j) {
    int k = lane + 32 * j;
    vals[j] = (k < D_) ? gload1(xr + k) : 0.f;
    s += vals[j];
  }
  #pragma unroll
  for (int o = 16; o > 0; o >>= 1) s += __shfl_xor(s, o, 32);
  float mu = s * (1.f / D_);
  float vs = 0.f;
  #pragma unroll
  for (int j = 0; j < 4; ++j) {
    int k = lane + 32 * j;
    float d = (k < D_) ? (vals[j] - mu) : 0.f;
    vs += d * d;
  }
  #pragma unroll
  for (int o = 16; o > 0; o >>= 1) vs += __shfl_xor(vs, o, 32);
  float inv = rsqrtf(vs * (1.f / D_) + 1e-6f);
  float* yr = y + wid * D_;
  #pragma unroll
  for (int j = 0; j < 4; ++j) {
    int k = lane + 32 * j;
    if (k < D_) gstore1(yr + k, (vals[j] - mu) * inv * g[k] + bb[k]);
  }
}

// Softmax scores over tiny sequence (S<=3), one thread per (group, qpos).
__global__ void attn_scores_kernel(const float* __restrict__ q, const float* __restrict__ k,
                                   float* __restrict__ attn, int G, int S)
{
  long long i = (long long)blockIdx.x * blockDim.x + threadIdx.x;
  if (i >= (long long)G * S) return;
  long long g = i / S; int qp = (int)(i - g * S);
  const float* qr = q + (g * S + qp) * D_;
  float sc[3];
  float mx = -1e30f;
  for (int s = 0; s < S; ++s) {
    const float* kr = k + (g * S + s) * D_;
    float d = 0.f;
    for (int c = 0; c < D_; ++c) d += gload1(qr + c) * gload1(kr + c);
    sc[s] = d * 0.1f;                    // 1/sqrt(D), D=100
    mx = fmaxf(mx, sc[s]);
  }
  float tot = 0.f;
  for (int s = 0; s < S; ++s) { sc[s] = expf(sc[s] - mx); tot += sc[s]; }
  float inv = 1.f / tot;
  for (int s = 0; s < S; ++s) gstore1(attn + i * S + s, sc[s] * inv);
}

__global__ void attn_apply_kernel(const float* __restrict__ attn, const float* __restrict__ v,
                                  float* __restrict__ av, int G, int S)
{
  long long i = (long long)blockIdx.x * blockDim.x + threadIdx.x;
  long long total = (long long)G * S * D_;
  if (i >= total) return;
  int d = (int)(i % D_);
  long long row = i / D_;
  long long g = row / S;
  float acc = 0.f;
  for (int s = 0; s < S; ++s) acc += gload1(attn + row * S + s) * gload1(v + (g * S + s) * D_ + d);
  gstore1(av + i, acc);
}

// Build mv [B,3,T,D] from spec0[t][B,D], spec1[t][B,D], rview[B,D].
__global__ void stack_mv_kernel(const float* __restrict__ spec0, const float* __restrict__ spec1,
                                const float* __restrict__ rview, float* __restrict__ mv)
{
  long long i = (long long)blockIdx.x * blockDim.x + threadIdx.x;
  const long long total = (long long)B_ * 3 * T_ * D_;
  if (i >= total) return;
  int d = (int)(i % D_);
  long long r = i / D_;
  int t = (int)(r % T_); r /= T_;
  int view = (int)(r % 3);
  long long b = r / 3;
  float v;
  if (view == 0)      v = gload1(spec0 + (long long)t * B_ * D_ + b * D_ + d);
  else if (view == 1) v = gload1(spec1 + (long long)t * B_ * D_ + b * D_ + d);
  else                v = gload1(rview + b * D_ + d);
  gstore1(mv + i, v);
}

// [B,3,T,D] -> [B,T,3,D]
__global__ void transpose_kernel(const float* __restrict__ src, float* __restrict__ dst)
{
  long long i = (long long)blockIdx.x * blockDim.x + threadIdx.x;
  const long long total = (long long)B_ * 3 * T_ * D_;
  if (i >= total) return;
  int d = (int)(i % D_);
  long long r = i / D_;
  int t = (int)(r % T_); r /= T_;
  int view = (int)(r % 3);
  long long b = r / 3;
  gstore1(dst + (((b * T_) + t) * 3 + view) * D_ + d, gload1(src + i));
}

// pooled[b,t,d] = mean over 3 views of mt[B,T,3,D]
__global__ void pool_kernel(const float* __restrict__ mt, float* __restrict__ pooled)
{
  long long i = (long long)blockIdx.x * blockDim.x + threadIdx.x;
  const long long total = (long long)B_ * T_ * D_;
  if (i >= total) return;
  int d = (int)(i % D_);
  long long r = i / D_;
  int t = (int)(r % T_);
  long long b = r / T_;
  float s = 0.f;
  for (int u = 0; u < 3; ++u) s += gload1(mt + (((b * T_) + t) * 3 + u) * D_ + d);
  gstore1(pooled + i, s * (1.f / 3.f));
}

// sel = pooled[b, et[b]]; out = normalize(base[node] + sel @ reflect[et[b]])
__global__ void final_kernel(const float* __restrict__ pooled, const int* __restrict__ edgetype,
                             const int* __restrict__ nodeids, const float* __restrict__ reflect,
                             const float* __restrict__ base, float* __restrict__ outp)
{
  __shared__ float sel[D_];
  __shared__ float red[8];
  int b = blockIdx.x;
  int t = edgetype[B_ + b];              // edgetype[1][b]; edgetype[0][b]==b
  for (int kk = threadIdx.x; kk < D_; kk += blockDim.x)
    sel[kk] = gload1(pooled + ((long long)b * T_ + t) * D_ + kk);
  __syncthreads();
  int o = threadIdx.x;
  float myval = 0.f;
  if (o < OUTD_) {
    const float* W = reflect + (long long)t * D_ * OUTD_;
    float acc = 0.f;
    for (int dd = 0; dd < D_; ++dd) acc += sel[dd] * gload1(W + dd * OUTD_ + o);
    myval = acc + gload1(base + (long long)nodeids[b] * OUTD_ + o);
  }
  float sq = myval * myval;
  #pragma unroll
  for (int off = 16; off > 0; off >>= 1) sq += __shfl_xor(sq, off, 32);
  if ((threadIdx.x & 31) == 0) red[threadIdx.x >> 5] = sq;
  __syncthreads();
  float tot = 0.f;
  for (int w = 0; w < 8; ++w) tot += red[w];
  float nrm = fmaxf(sqrtf(tot), 1e-12f);
  if (o < OUTD_) gstore1(outp + (long long)b * OUTD_ + o, myval / nrm);
}

// ---------------------------------------------------------------------------
// Host-side helpers
// ---------------------------------------------------------------------------
static inline GSrc srcDense(const float* p, int r) {
  GSrc s{}; s.dense = p; s.r = r; s.P = 1; return s;
}
static inline GSrc srcGather(const float* table, long long rowstride, long long coloff,
                             const int* nb, int bstride, int toff, int P, int r) {
  GSrc s{}; s.table = table; s.rowstride = rowstride; s.coloff = coloff;
  s.nb = nb; s.bstride = bstride; s.toff = toff; s.P = P; s.r = r; return s;
}

static void launch_gemm(const GSrc& X, const GSrc& Y, int ry,
                        const float* W0, int N0, const float* W1, int N1,
                        float* out, const float* resid, long long n, int relu,
                        hipStream_t st)
{
  unsigned blocks = (unsigned)((n + 63) / 64);   // 4 waves x 16 rows per block
  int KP = (ry > 0) ? 256 : 128;
  size_t lds = (size_t)NCOLS_ * (KP + 8) * sizeof(_Float16);
  switch (ry) {
    case 0: wmma_gemm_kernel<0><<<dim3(blocks), dim3(128), lds, st>>>(X, Y, W0, N0, W1, N1, out, resid, n, relu); break;
    case 3: wmma_gemm_kernel<3><<<dim3(blocks), dim3(128), lds, st>>>(X, Y, W0, N0, W1, N1, out, resid, n, relu); break;
    case 5: wmma_gemm_kernel<5><<<dim3(blocks), dim3(128), lds, st>>>(X, Y, W0, N0, W1, N1, out, resid, n, relu); break;
    case 7: wmma_gemm_kernel<7><<<dim3(blocks), dim3(128), lds, st>>>(X, Y, W0, N0, W1, N1, out, resid, n, relu); break;
    case 9: wmma_gemm_kernel<9><<<dim3(blocks), dim3(128), lds, st>>>(X, Y, W0, N0, W1, N1, out, resid, n, relu); break;
    default: break;
  }
}

// One GraphSAGE pyramid (schema view or random-walk view) for one (table, t).
static void run_pyramid(const int* parts, int L,
                        const int* nb, int bstride, int toffbase,
                        const int* nodeids,
                        const float* table, long long rowstride, long long coloff,
                        const float* Wself, const float* Wneigh,   // [L][100][50]
                        float* gA, float* gB, float* outP, hipStream_t st)
{
  long long rows[5];
  rows[0] = B_;
  for (int k = 1; k <= L; ++k) rows[k] = (long long)B_ * parts[k - 1];
  int fan[4];
  fan[0] = parts[0];
  for (int k = 1; k < L; ++k) fan[k] = parts[k] / parts[k - 1];
  long long sub[5];
  sub[0] = 0;
  for (int k = 0; k < L; ++k) sub[k + 1] = sub[k] + rows[k] * D_;
  int coff[5];
  coff[0] = 0;
  for (int k = 0; k < L; ++k) coff[k + 1] = coff[k] + parts[k];

  // Level 0: X and Y both gathered from the embedding table (fused in GEMM).
  for (int k = 0; k < L; ++k) {
    GSrc X = (k == 0)
      ? srcGather(table, rowstride, coloff, nodeids, 1, 0, 1, 1)
      : srcGather(table, rowstride, coloff, nb, bstride, toffbase + coff[k - 1],
                  parts[k - 1], 1);
    GSrc Y = srcGather(table, rowstride, coloff, nb, bstride, toffbase + coff[k],
                       (k == 0) ? 1 : parts[k - 1], fan[k]);
    launch_gemm(X, Y, fan[k], Wself, 50, Wneigh, 50, gA + sub[k], nullptr, rows[k], 1, st);
  }
  // Levels >= 1: dense inputs, ping-pong arenas.
  float* prev = gA; float* cur = gB;
  for (int l = 1; l < L; ++l) {
    for (int k = 0; k < L - l; ++k) {
      GSrc X = srcDense(prev + sub[k], 1);
      GSrc Y = srcDense(prev + sub[k + 1], fan[k]);
      float* o = (l == L - 1) ? outP : (cur + sub[k]);
      launch_gemm(X, Y, fan[k], Wself + (long long)l * D_ * 50,
                  50, Wneigh + (long long)l * D_ * 50, 50,
                  o, nullptr, rows[k], 1, st);
    }
    float* tp = prev; prev = cur; cur = tp;
  }
}

static void run_mha(const float* x, float* out, int G, int S,
                    const float* Wq, const float* Wk, const float* Wv, const float* Wf,
                    const float* g, const float* bb,
                    float* xln, float* q, float* k, float* v, float* attn, float* av,
                    hipStream_t st)
{
  long long rows = (long long)G * S;
  GSrc none{};
  ln_kernel<<<(unsigned)((rows + 3) / 4), 128, 0, st>>>(x, xln, g, bb, rows);
  launch_gemm(srcDense(xln, 1), none, 0, Wq, D_, nullptr, 0, q, nullptr, rows, 0, st);
  launch_gemm(srcDense(x, 1),   none, 0, Wk, D_, nullptr, 0, k, nullptr, rows, 0, st);
  launch_gemm(srcDense(x, 1),   none, 0, Wv, D_, nullptr, 0, v, nullptr, rows, 0, st);
  attn_scores_kernel<<<(unsigned)((rows + 127) / 128), 128, 0, st>>>(q, k, attn, G, S);
  long long tot = rows * D_;
  attn_apply_kernel<<<(unsigned)((tot + 255) / 256), 256, 0, st>>>(attn, v, av, G, S);
  launch_gemm(srcDense(av, 1), none, 0, Wf, D_, nullptr, 0, out, x /*residual*/, rows, 0, st);
}

// ---------------------------------------------------------------------------
extern "C" void kernel_launch(void* const* d_in, const int* in_sizes, int n_in,
                              void* d_out, int out_size, void* d_ws, size_t ws_size,
                              hipStream_t stream)
{
  (void)in_sizes; (void)n_in; (void)out_size; (void)ws_size;
  const int* nodeids    = (const int*)d_in[0];
  const int* edgetype   = (const int*)d_in[1];
  const int* neighbors  = (const int*)d_in[2];
  const int* rnd        = (const int*)d_in[3];
  const float* base_embed = (const float*)d_in[4];
  const float* type_embed = (const float*)d_in[5];
  const float* rw_embed   = (const float*)d_in[6];
  const float* reflect    = (const float*)d_in[7];
  const float* agg0_self  = (const float*)d_in[8];
  const float* agg0_neigh = (const float*)d_in[9];
  const float* agg1_self  = (const float*)d_in[10];
  const float* agg1_neigh = (const float*)d_in[11];
  const float* rand_self  = (const float*)d_in[12];
  const float* rand_neigh = (const float*)d_in[13];
  const float* vWq = (const float*)d_in[14];
  const float* vWk = (const float*)d_in[15];
  const float* vWv = (const float*)d_in[16];
  const float* vWf = (const float*)d_in[17];
  const float* vg  = (const float*)d_in[18];
  const float* vb  = (const float*)d_in[19];
  const float* mWq = (const float*)d_in[20];
  const float* mWk = (const float*)d_in[21];
  const float* mWv = (const float*)d_in[22];
  const float* mWf = (const float*)d_in[23];
  const float* mg  = (const float*)d_in[24];
  const float* mb  = (const float*)d_in[25];
  float* outp = (float*)d_out;

  // Workspace allocator (~42 MB total).
  char* wbase = (char*)d_ws;
  size_t woff = 0;
  auto alloc = [&](long long nf) -> float* {
    float* p = (float*)(wbase + woff);
    woff += (size_t)nf * sizeof(float);
    woff = (woff + 255) & ~(size_t)255;
    return p;
  };
  float* gA    = alloc(124LL * B_ * D_);   // level-0 pyramid outputs (schema1 worst case)
  float* gB    = alloc(19LL * B_ * D_);    // level-1 outputs
  float* spec0 = alloc(2LL * B_ * D_);
  float* spec1 = alloc(2LL * B_ * D_);
  float* rview = alloc((long long)B_ * D_);
  const long long mvN = (long long)B_ * 3 * T_ * D_;
  float* mv   = alloc(mvN);
  float* mvT  = alloc(mvN);
  float* mvo  = alloc(mvN);
  float* mt   = alloc(mvN);
  float* xln  = alloc(mvN);
  float* qb   = alloc(mvN);
  float* kb   = alloc(mvN);
  float* vbuf = alloc(mvN);
  float* av   = alloc(mvN);
  float* attn = alloc(16384);
  float* pooled = alloc((long long)B_ * T_ * D_);

  static const int parts0[2] = {3, 15};
  static const int parts1[4] = {3, 15, 105, 945};

  // Schema views: type_embed is [MAX_USERS, T, 2, D] -> rowstride 400,
  // column offset (t*2 + schema)*100.  neighbors is [B, T, 1086].
  for (int t = 0; t < T_; ++t) {
    run_pyramid(parts0, 2, neighbors, T_ * TOTALP_, t * TOTALP_ + 0, nodeids,
                type_embed, (long long)T_ * 2 * D_, (long long)(t * 2 + 0) * D_,
                agg0_self, agg0_neigh, gA, gB,
                spec0 + (long long)t * B_ * D_, stream);
    run_pyramid(parts1, 4, neighbors, T_ * TOTALP_, t * TOTALP_ + 18, nodeids,
                type_embed, (long long)T_ * 2 * D_, (long long)(t * 2 + 1) * D_,
                agg1_self, agg1_neigh, gA, gB,
                spec1 + (long long)t * B_ * D_, stream);
  }
  // Random-walk view: rw_embed [MAX_USERS, D].
  run_pyramid(parts0, 2, rnd, 18, 0, nodeids, rw_embed, D_, 0,
              rand_self, rand_neigh, gA, gB, rview, stream);

  // mv [B,3,T,D] -> view attention (over T) -> meta attention (over views).
  stack_mv_kernel<<<(unsigned)((mvN + 255) / 256), 256, 0, stream>>>(spec0, spec1, rview, mv);
  run_mha(mv, mvo, B_ * 3, T_, vWq, vWk, vWv, vWf, vg, vb,
          xln, qb, kb, vbuf, attn, av, stream);
  transpose_kernel<<<(unsigned)((mvN + 255) / 256), 256, 0, stream>>>(mvo, mvT);
  run_mha(mvT, mt, B_ * T_, 3, mWq, mWk, mWv, mWf, mg, mb,
          xln, qb, kb, vbuf, attn, av, stream);

  pool_kernel<<<(unsigned)(((long long)B_ * T_ * D_ + 255) / 256), 256, 0, stream>>>(mt, pooled);
  final_kernel<<<B_, 256, 0, stream>>>(pooled, edgetype, nodeids, reflect, base_embed, outp);
}
